// PointNetFeaturePropagation_31980326486609
// MI455X (gfx1250) — compile-verified, matrix operand order checked by reading, not run on hardware
//
#include <hip/hip_runtime.h>

// Problem constants (from reference setup_inputs)
#define BB   16
#define NN   4096
#define SS   1024
#define D1C  64
#define D2C  256
#define CIN  320     // D1 + D2
#define HO   256     // hidden/output channels
#define MTOT (BB * NN)   // 65536 rows

typedef __attribute__((ext_vector_type(16))) __bf16 v16bf;
typedef __attribute__((ext_vector_type(8)))  float  v8f;

union Frag { unsigned int u[8]; v16bf b; };

static __device__ __forceinline__ unsigned short f32_to_bf16(float f) {
    unsigned int u = __float_as_uint(f);
    // round-to-nearest-even
    u += 0x7FFFu + ((u >> 16) & 1u);
    return (unsigned short)(u >> 16);
}

// ---------------------------------------------------------------------------
// Kernel 1: convert W1 [256x320] and W2 [256x256] f32 -> bf16 (row-major kept)
// ---------------------------------------------------------------------------
__global__ void cvt_w_kernel(const float* __restrict__ W1, const float* __restrict__ W2,
                             unsigned short* __restrict__ Wb1, unsigned short* __restrict__ Wb2) {
    int i = blockIdx.x * 256 + threadIdx.x;
    if (i < HO * CIN) Wb1[i] = f32_to_bf16(W1[i]);
    if (i < HO * HO)  Wb2[i] = f32_to_bf16(W2[i]);
}

// ---------------------------------------------------------------------------
// Kernel 2: 3-NN inverse-distance interpolation + concat -> X bf16 [M x 320]
// One block = 256 points of one batch. xyz2[b] (12 KB) staged in LDS.
// ---------------------------------------------------------------------------
__global__ void __launch_bounds__(256)
interp_concat_kernel(const float* __restrict__ xyz1, const float* __restrict__ xyz2,
                     const float* __restrict__ p1,   const float* __restrict__ p2,
                     unsigned short* __restrict__ X) {
    __shared__ float lx[SS], ly[SS], lz[SS];
    const int b = blockIdx.y;
    const int n = blockIdx.x * 256 + threadIdx.x;

    const float* x2b = xyz2 + (size_t)b * 3 * SS;
    for (int s = threadIdx.x; s < SS; s += 256) {
        lx[s] = x2b[s];
        ly[s] = x2b[SS + s];
        lz[s] = x2b[2 * SS + s];
    }
    __syncthreads();

    const float* x1b = xyz1 + (size_t)b * 3 * NN;
    const float px = x1b[n], py = x1b[NN + n], pz = x1b[2 * NN + n];

    float b0 = 3.402823e38f, b1v = 3.402823e38f, b2v = 3.402823e38f;
    int   i0 = 0, i1 = 0, i2 = 0;
    for (int s = 0; s < SS; ++s) {
        const float dx = px - lx[s], dy = py - ly[s], dz = pz - lz[s];
        const float d = dx * dx + dy * dy + dz * dz;
        if (d < b0)       { b2v = b1v; i2 = i1; b1v = b0; i1 = i0; b0 = d; i0 = s; }
        else if (d < b1v) { b2v = b1v; i2 = i1; b1v = d;  i1 = s; }
        else if (d < b2v) { b2v = d;  i2 = s; }
    }
    const float r0 = 1.0f / (b0 + 1e-8f);
    const float r1 = 1.0f / (b1v + 1e-8f);
    const float r2 = 1.0f / (b2v + 1e-8f);
    const float inv = 1.0f / (r0 + r1 + r2);
    const float w0 = r0 * inv, w1 = r1 * inv, w2 = r2 * inv;

    unsigned short* Xr = X + (size_t)(b * NN + n) * CIN;

    // points1 slice (coalesced across threads for each c)
    const float* p1b = p1 + (size_t)b * D1C * NN + n;
    #pragma unroll 4
    for (int c = 0; c < D1C; ++c)
        Xr[c] = f32_to_bf16(p1b[(size_t)c * NN]);

    // interpolated points2 features
    const float* p2b = p2 + (size_t)b * D2C * SS;
    for (int d = 0; d < D2C; ++d) {
        const float* row = p2b + (size_t)d * SS;
        const float v = w0 * row[i0] + w1 * row[i1] + w2 * row[i2];
        Xr[D1C + d] = f32_to_bf16(v);
    }
}

// ---------------------------------------------------------------------------
// GEMM + bias + BN-stat accumulation.
// A: [M x K] bf16 row-major, Bw: [256 x K] bf16 row-major (weights).
// Each wave: 16 rows x 256 cols via 16 v_wmma_f32_16x16x32_bf16 accumulators.
// Fragment layouts follow CDNA5 ISA 7.12.2 (16-bit A 16x32 / B 32x16 K-pair
// striping: lane half selects K half, VGPR v -> K = 16*(v>=4) + 8*half + 2*(v&3)).
// A/B fragments are ping-pong double-buffered: since VMEM loads return in
// order, the next tile's two b128 loads stay in flight behind the current
// WMMA instead of a full s_wait_loadcnt 0 before every matrix op.
// Epilogue: H[row][o] = acc + bias[o]; per-channel sum/sumsq via shfl+atomics.
// ---------------------------------------------------------------------------
template <int K>
__global__ void __launch_bounds__(128)
gemm_bias_stats_kernel(const unsigned short* __restrict__ A,
                       const unsigned short* __restrict__ Bw,
                       const float* __restrict__ bias,
                       float* __restrict__ H,
                       float* __restrict__ sum, float* __restrict__ sq) {
    const int lane = threadIdx.x & 31;
    const int wave = threadIdx.x >> 5;
    const int rbase = (blockIdx.x * 4 + wave) * 16;
    const int half = lane >> 4;       // K-half selector
    const int mn   = lane & 15;       // row (A) / col (B) within tile

    // per-VGPR K offsets in uint (bf16-pair) units:
    //   uints {0,1,2,3} and {8,9,10,11} relative to (half*4 + kb/2)
    int ko[8];
    #pragma unroll
    for (int v = 0; v < 8; ++v)
        ko[v] = ((v & 4) ? 8 : 0) + half * 4 + (v & 3);

    const unsigned int* Arow = (const unsigned int*)(A + (size_t)(rbase + mn) * K);
    const unsigned int* Brow = (const unsigned int*)(Bw + (size_t)mn * K);

    auto loadA = [&](Frag& f, int kb) {
        const unsigned int* p = Arow + (kb >> 1);
        #pragma unroll
        for (int v = 0; v < 8; ++v) f.u[v] = p[ko[v]];
    };
    auto loadB = [&](Frag& f, int t, int kb) {
        const unsigned int* p = Brow + (size_t)t * 16 * K / 2 + (kb >> 1);
        #pragma unroll
        for (int v = 0; v < 8; ++v) f.u[v] = p[ko[v]];
    };

    v8f acc[16];
    #pragma unroll
    for (int t = 0; t < 16; ++t)
        #pragma unroll
        for (int e = 0; e < 8; ++e) acc[t][e] = 0.0f;

    constexpr int NK = K / 32;
    Frag fa[2], fb[2];
    loadA(fa[0], 0);
    loadB(fb[0], 0, 0);

    #pragma unroll
    for (int ks = 0; ks < NK; ++ks) {
        const int kb = ks * 32;
        if (ks + 1 < NK) loadA(fa[(ks + 1) & 1], kb + 32);
        #pragma unroll
        for (int t = 0; t < 16; ++t) {
            if (t < 15)             loadB(fb[(t + 1) & 1], t + 1, kb);
            else if (ks + 1 < NK)   loadB(fb[0],           0,     kb + 32);
            acc[t] = __builtin_amdgcn_wmma_f32_16x16x32_bf16(
                false, fa[ks & 1].b, false, fb[t & 1].b, (short)0, acc[t], false, false);
        }
    }

    // C/D layout: lane L, vgpr r -> row = rbase + 8*half + r, col = mn (+16*t)
    const int rtop = rbase + half * 8;
    #pragma unroll
    for (int t = 0; t < 16; ++t) {
        const int o = t * 16 + mn;
        const float bv = bias[o];
        float s = 0.0f, q = 0.0f;
        #pragma unroll
        for (int r = 0; r < 8; ++r) {
            const float val = acc[t][r] + bv;
            H[(size_t)(rtop + r) * HO + o] = val;
            s += val;
            q += val * val;
        }
        // lane L and L^16 hold the same column -> combine halves
        s += __shfl_xor(s, 16, 32);
        q += __shfl_xor(q, 16, 32);
        if (half == 0) {
            atomicAdd(&sum[o], s);
            atomicAdd(&sq[o], q);
        }
    }
}

// ---------------------------------------------------------------------------
// BN finalize: scale = g * rsqrt(var + eps); shift = bt - mean*scale
// (biased variance, matching jnp.var)
// ---------------------------------------------------------------------------
__global__ void bn_finalize_kernel(const float* __restrict__ sum, const float* __restrict__ sq,
                                   const float* __restrict__ g,   const float* __restrict__ bt,
                                   float* __restrict__ scale, float* __restrict__ shift,
                                   float inv_count) {
    const int c = threadIdx.x;
    const float m = sum[c] * inv_count;
    const float v = fmaxf(sq[c] * inv_count - m * m, 0.0f);
    const float sc = g[c] * rsqrtf(v + 1e-5f);
    scale[c] = sc;
    shift[c] = fmaf(-m, sc, bt[c]);
}

// ---------------------------------------------------------------------------
// BN + ReLU -> bf16 activations for the next GEMM  (H [M x 256] -> X2 bf16)
// ---------------------------------------------------------------------------
__global__ void __launch_bounds__(256)
bn_relu_bf16_kernel(const float* __restrict__ H,
                    const float* __restrict__ scale, const float* __restrict__ shift,
                    unsigned short* __restrict__ X2) {
    const size_t i = (size_t)blockIdx.x * 256 + threadIdx.x;
    const int c = (int)(i & (HO - 1));
    const float y = fmaxf(fmaf(H[i], scale[c], shift[c]), 0.0f);
    X2[i] = f32_to_bf16(y);
}

// ---------------------------------------------------------------------------
// Final BN + ReLU + transpose to [B, 256, N] f32 output.
// LDS tile 32 n x 256 o, 257-padded: coalesced load and store, conflict-free.
// ---------------------------------------------------------------------------
__global__ void __launch_bounds__(256)
bn_relu_out_kernel(const float* __restrict__ H,
                   const float* __restrict__ scale, const float* __restrict__ shift,
                   float* __restrict__ out) {
    __shared__ float tile[32][HO + 1];
    const int b  = blockIdx.y;
    const int n0 = blockIdx.x * 32;
    const int tid = threadIdx.x;

    #pragma unroll 4
    for (int j = 0; j < 32; ++j)
        tile[j][tid] = H[((size_t)b * NN + n0 + j) * HO + tid];
    __syncthreads();

    #pragma unroll 4
    for (int j = 0; j < 32; ++j) {
        const int o = j * 8 + (tid >> 5);
        const int n = n0 + (tid & 31);
        const float v = fmaxf(fmaf(tile[tid & 31][o], scale[o], shift[o]), 0.0f);
        out[((size_t)b * HO + o) * NN + n] = v;
    }
}

// ---------------------------------------------------------------------------
// Host-side orchestration
// ---------------------------------------------------------------------------
extern "C" void kernel_launch(void* const* d_in, const int* in_sizes, int n_in,
                              void* d_out, int out_size, void* d_ws, size_t ws_size,
                              hipStream_t stream) {
    const float* xyz1 = (const float*)d_in[0];
    const float* xyz2 = (const float*)d_in[1];
    const float* pts1 = (const float*)d_in[2];
    const float* pts2 = (const float*)d_in[3];
    const float* W1   = (const float*)d_in[4];
    const float* b1   = (const float*)d_in[5];
    const float* g1   = (const float*)d_in[6];
    const float* bt1  = (const float*)d_in[7];
    const float* W2   = (const float*)d_in[8];
    const float* b2   = (const float*)d_in[9];
    const float* g2   = (const float*)d_in[10];
    const float* bt2  = (const float*)d_in[11];
    float* out = (float*)d_out;

    // workspace layout (peak ~109 MiB):
    //   [0, 40MiB)      : X bf16 [65536 x 320]  -> later reused as X2 bf16 [65536 x 256]
    //   [40, 104MiB)    : H f32  [65536 x 256]  (H1, then reused for H2)
    //   then weights bf16 + stats
    char* ws = (char*)d_ws;
    unsigned short* Xb  = (unsigned short*)ws;                               // 41,943,040 B
    float*          Hb  = (float*)(ws + (size_t)MTOT * CIN * 2);             // 67,108,864 B
    unsigned short* Wb1 = (unsigned short*)((char*)Hb + (size_t)MTOT * HO * 4);
    unsigned short* Wb2 = Wb1 + HO * CIN;
    float* stats = (float*)(Wb2 + HO * HO);
    float* sum1   = stats + 0 * HO;
    float* sq1    = stats + 1 * HO;
    float* sum2   = stats + 2 * HO;
    float* sq2    = stats + 3 * HO;
    float* scale1 = stats + 4 * HO;
    float* shift1 = stats + 5 * HO;
    float* scale2 = stats + 6 * HO;
    float* shift2 = stats + 7 * HO;
    unsigned short* X2b = Xb;   // reuse X region after GEMM1

    // zero the four stat accumulators (graph-capture safe)
    hipMemsetAsync(stats, 0, 4 * HO * sizeof(float), stream);

    // 1) weights -> bf16
    cvt_w_kernel<<<(HO * CIN + 255) / 256, 256, 0, stream>>>(W1, W2, Wb1, Wb2);

    // 2) 3-NN interpolation + concat -> X bf16
    interp_concat_kernel<<<dim3(NN / 256, BB), 256, 0, stream>>>(xyz1, xyz2, pts1, pts2, Xb);

    // 3) GEMM1 (K=320) + bias + BN stats
    gemm_bias_stats_kernel<CIN><<<MTOT / 64, 128, 0, stream>>>(Xb, Wb1, b1, Hb, sum1, sq1);

    // 4) BN1 finalize
    bn_finalize_kernel<<<1, HO, 0, stream>>>(sum1, sq1, g1, bt1, scale1, shift1,
                                             1.0f / (float)MTOT);

    // 5) BN1 + ReLU -> X2 bf16
    bn_relu_bf16_kernel<<<(size_t)MTOT * HO / 256, 256, 0, stream>>>(Hb, scale1, shift1, X2b);

    // 6) GEMM2 (K=256) + bias + BN stats (H region reused)
    gemm_bias_stats_kernel<HO><<<MTOT / 64, 128, 0, stream>>>(X2b, Wb2, b2, Hb, sum2, sq2);

    // 7) BN2 finalize
    bn_finalize_kernel<<<1, HO, 0, stream>>>(sum2, sq2, g2, bt2, scale2, shift2,
                                             1.0f / (float)MTOT);

    // 8) BN2 + ReLU + transpose to [B, 256, N]
    bn_relu_out_kernel<<<dim3(NN / 32, BB), 256, 0, stream>>>(Hb, scale2, shift2, out);
}
